// Decoder_32830730011582
// MI455X (gfx1250) — compile-verified, matrix operand order checked by reading, not run on hardware
//
#include <hip/hip_runtime.h>

// ---------------------------------------------------------------------------
// out[b, i, a] = sum_{l<=i} sum_f x[b, l, f] * W_i[l, f, a]
//   B=128, L=12, F=4096, A=768. f32 in/out.
// Per-workgroup: 128(M) x 64(N) x 4096(K) partial GEMM, split-precision bf16
// WMMA (hi+lo truncation split, 3 wmmas/product), double-buffered LDS with
// register pre-staging, accumulated via global_atomic_add_f32.
// Grid: 12 N-tiles x 78 (layer,slice) pairs = 936 workgroups.
// ---------------------------------------------------------------------------

#define BATCH   128
#define NLAYERS 12
#define DFEAT   4096
#define DACT    768
#define KTILE   32
#define NTILE   64
#define LDSTR   40              // padded LDS row stride (bf16): 80B, 16B-aligned
#define XBUF    (BATCH * LDSTR) // 5120 elems per x buffer
#define WBUF    (NTILE * LDSTR) // 2560 elems per w buffer

typedef __attribute__((ext_vector_type(8)))  float          v8f;
typedef __attribute__((ext_vector_type(16))) __bf16         v16bf;
typedef __attribute__((ext_vector_type(4)))  unsigned short us4;
typedef __attribute__((ext_vector_type(8)))  unsigned short us8;
typedef __attribute__((ext_vector_type(16))) unsigned short us16;

struct WArr { const float* w[NLAYERS]; };

// Truncation split: f ~= hi + lo with hi = top 16 bits, lo = bf16(f - hi).
__device__ __forceinline__ unsigned short bf16_hi(unsigned int u) {
    return (unsigned short)(u >> 16);
}
__device__ __forceinline__ float hi_as_f32(unsigned int u) {
    return __uint_as_float(u & 0xFFFF0000u);
}
__device__ __forceinline__ unsigned short bf16_lo(float f, unsigned int u) {
    return (unsigned short)(__float_as_uint(f - hi_as_f32(u)) >> 16);
}

// 16x32 bf16 WMMA operand fragment from LDS (K-contiguous rows).
// lanes 0-15: K=0..7 (v0-3), K=16..23 (v4-7); lanes 16-31: K=8..15, K=24..31.
__device__ __forceinline__ v16bf load_frag(const unsigned short* lds, int base) {
    us8 a = *(const us8*)(lds + base);
    us8 b = *(const us8*)(lds + base + 16);
    us16 r = __builtin_shufflevector(a, b, 0, 1, 2, 3, 4, 5, 6, 7,
                                           8, 9, 10, 11, 12, 13, 14, 15);
    return __builtin_bit_cast(v16bf, r);
}

__global__ void __launch_bounds__(256)
decoder_zero_kernel(float* out, int n) {
    int idx = blockIdx.x * blockDim.x + threadIdx.x;
    if (idx < n) out[idx] = 0.0f;
}

__global__ void __launch_bounds__(256)
decoder_gemm_kernel(const float* __restrict__ x, WArr wa, float* __restrict__ out) {
    const int n0 = blockIdx.x * NTILE;            // 12 N-tiles of 64
    int layer = 0, l = blockIdx.y;                // 78 (i, l) pairs
    while (l >= layer + 1) { l -= layer + 1; ++layer; }

    const float* Wp = wa.w[0];
#pragma unroll
    for (int j = 1; j < NLAYERS; ++j)
        if (layer == j) Wp = wa.w[j];
    const float* __restrict__ Wl = Wp + (size_t)l * DFEAT * DACT;   // [4096, 768]
    const float* __restrict__ xl = x + (size_t)l * DFEAT;

    __shared__ unsigned short xs_hi[2 * XBUF];    // double-buffered: 60 KB total
    __shared__ unsigned short xs_lo[2 * XBUF];
    __shared__ unsigned short ws_hi[2 * WBUF];
    __shared__ unsigned short ws_lo[2 * WBUF];

    const int tid  = threadIdx.x;
    const int lane = tid & 31;
    const int wv   = tid >> 5;
    const int wm   = wv & 3;                      // 4 wave-rows of M=32
    const int wnb  = (wv >> 2) * 32;              // 2 wave-cols of N=32
    const int mr0  = wm * 32 + (lane & 15);
    const int mr1  = mr0 + 16;
    const int koff = (lane >> 4) * 8;

    // ---- staging helpers: 4 float4 (x) + 2 float4 (W) per thread/chunk ----
    auto stage_load = [&](int kc, float4* xr, float4* wr) {
#pragma unroll
        for (int it = 0; it < 4; ++it) {
            int p = tid + it * 256;               // 1024 float4: 128 rows x 8
            xr[it] = *(const float4*)(xl + (size_t)(p >> 3) * (NLAYERS * DFEAT)
                                         + kc + (p & 7) * 4);
        }
#pragma unroll
        for (int it = 0; it < 2; ++it) {
            int p = tid + it * 256;               // 512 float4: 32 rows x 16
            const float* wp = Wl + (size_t)(kc + (p >> 4)) * DACT + n0 + (p & 15) * 4;
            if (kc + KTILE < DFEAT)
                __builtin_prefetch(wp + KTILE * DACT, 0, 0);   // global_prefetch_b8
            wr[it] = *(const float4*)wp;
        }
    };
    auto stage_store = [&](const float4* xr, const float4* wr, int buf) {
        const int box = buf * XBUF, bow = buf * WBUF;
#pragma unroll
        for (int it = 0; it < 4; ++it) {
            int p = tid + it * 256;
            int m = p >> 3, c4 = (p & 7) * 4;
            const float4 v = xr[it];
            unsigned int u0 = __float_as_uint(v.x), u1 = __float_as_uint(v.y),
                         u2 = __float_as_uint(v.z), u3 = __float_as_uint(v.w);
            us4 hi = {bf16_hi(u0), bf16_hi(u1), bf16_hi(u2), bf16_hi(u3)};
            us4 lo = {bf16_lo(v.x, u0), bf16_lo(v.y, u1),
                      bf16_lo(v.z, u2), bf16_lo(v.w, u3)};
            *(us4*)(xs_hi + box + m * LDSTR + c4) = hi;
            *(us4*)(xs_lo + box + m * LDSTR + c4) = lo;
        }
#pragma unroll
        for (int it = 0; it < 2; ++it) {
            int p = tid + it * 256;
            int k = p >> 4, col4 = (p & 15) * 4;
            const float4 v = wr[it];
            const float e[4] = {v.x, v.y, v.z, v.w};
#pragma unroll
            for (int c = 0; c < 4; ++c) {
                unsigned int u = __float_as_uint(e[c]);
                ws_hi[bow + (col4 + c) * LDSTR + k] = bf16_hi(u);        // transposed
                ws_lo[bow + (col4 + c) * LDSTR + k] = bf16_lo(e[c], u);
            }
        }
    };

    v8f acc[2][2] = {};                           // 32(M) x 32(N) per wave

    float4 xrg[4], wrg[2];
    stage_load(0, xrg, wrg);
    stage_store(xrg, wrg, 0);
    __syncthreads();

    const int nch = DFEAT / KTILE;                // 128 chunks
    for (int c = 0; c < nch; ++c) {
        const bool more = (c + 1 < nch);
        if (more) stage_load((c + 1) * KTILE, xrg, wrg);   // overlap with compute

        const int box = (c & 1) * XBUF, bow = (c & 1) * WBUF;
        const v16bf a_hi0 = load_frag(xs_hi + box, mr0 * LDSTR + koff);
        const v16bf a_lo0 = load_frag(xs_lo + box, mr0 * LDSTR + koff);
        const v16bf a_hi1 = load_frag(xs_hi + box, mr1 * LDSTR + koff);
        const v16bf a_lo1 = load_frag(xs_lo + box, mr1 * LDSTR + koff);
#pragma unroll
        for (int t = 0; t < 2; ++t) {
            const int nrow = wnb + t * 16 + (lane & 15);
            const v16bf b_hi = load_frag(ws_hi + bow, nrow * LDSTR + koff);
            const v16bf b_lo = load_frag(ws_lo + bow, nrow * LDSTR + koff);
            acc[0][t] = __builtin_amdgcn_wmma_f32_16x16x32_bf16(
                false, a_hi0, false, b_hi, (short)0, acc[0][t], false, false);
            acc[0][t] = __builtin_amdgcn_wmma_f32_16x16x32_bf16(
                false, a_hi0, false, b_lo, (short)0, acc[0][t], false, false);
            acc[0][t] = __builtin_amdgcn_wmma_f32_16x16x32_bf16(
                false, a_lo0, false, b_hi, (short)0, acc[0][t], false, false);
            acc[1][t] = __builtin_amdgcn_wmma_f32_16x16x32_bf16(
                false, a_hi1, false, b_hi, (short)0, acc[1][t], false, false);
            acc[1][t] = __builtin_amdgcn_wmma_f32_16x16x32_bf16(
                false, a_hi1, false, b_lo, (short)0, acc[1][t], false, false);
            acc[1][t] = __builtin_amdgcn_wmma_f32_16x16x32_bf16(
                false, a_lo1, false, b_hi, (short)0, acc[1][t], false, false);
        }

        if (more) stage_store(xrg, wrg, (c + 1) & 1);
        __syncthreads();    // separates reads of buf(c&1) from its next overwrite
    }

    // ---- accumulate into out[b, layer, a] ----
    // C layout: vgpr r -> row r (lanes 0-15) / r+8 (lanes 16-31); col = lane%16.
    const int rsub = (lane >> 4) << 3;
    const int acol = n0 + wnb + (lane & 15);
#pragma unroll
    for (int mi = 0; mi < 2; ++mi) {
#pragma unroll
        for (int t = 0; t < 2; ++t) {
#pragma unroll
            for (int r = 0; r < 8; ++r) {
                const int b = wm * 32 + mi * 16 + rsub + r;
                const int a = acol + t * 16;
                unsafeAtomicAdd(out + (size_t)b * (NLAYERS * DACT)
                                    + (size_t)layer * DACT + a, acc[mi][t][r]);
            }
        }
    }
}

extern "C" void kernel_launch(void* const* d_in, const int* in_sizes, int n_in,
                              void* d_out, int out_size, void* d_ws, size_t ws_size,
                              hipStream_t stream) {
    const float* x = (const float*)d_in[0];
    WArr wa;
    for (int j = 0; j < NLAYERS; ++j) wa.w[j] = (const float*)d_in[1 + j];
    float* out = (float*)d_out;

    decoder_zero_kernel<<<(out_size + 255) / 256, 256, 0, stream>>>(out, out_size);

    dim3 grid(DACT / NTILE, (NLAYERS * (NLAYERS + 1)) / 2);   // 12 x 78 = 936 WGs
    decoder_gemm_kernel<<<grid, 256, 0, stream>>>(x, wa, out);
}
